// FWDense_24223615549776
// MI455X (gfx1250) — compile-verified, matrix operand order checked by reading, not run on hardware
//
#include <hip/hip_runtime.h>
#include <hip/hip_bf16.h>

// ---------------------------------------------------------------------------
// Problem constants (from reference): B=4, L=4096, D=S=E=1024, 32 chunks of 128
// ---------------------------------------------------------------------------
#define B_ 4
#define L_ 4096
#define D_ 1024
#define S_ 1024
#define G_ 32
#define C_ 128

typedef __bf16 bf16_t;
typedef __attribute__((ext_vector_type(16))) __bf16 v16bf;
typedef __attribute__((ext_vector_type(8)))  float  v8f;

// LDS tile pitches (elements). KP = 32 K + 8 pad -> 80B rows: 16B aligned,
// conflict-free for the 2 x ds_load_b128 per-lane fragment fetch.
#define KP  40
#define QKP 136   // 128 + 8 pad -> 272B rows (16B aligned)

__device__ __forceinline__ v8f v8f_zero() {
  v8f z = {0.f, 0.f, 0.f, 0.f, 0.f, 0.f, 0.f, 0.f};
  return z;
}

// Fragment fetch for 16-bit A/B operands of v_wmma_f32_16x16x32_bf16.
// ISA 7.12.2: lane%16 = M (or N), lane/16 selects K-half; VGPR i holds K pair
// {2*(i%4), 2*(i%4)+1} + 8*(lane>>4) + 16*(i/4)  -> two contiguous 16B runs.
__device__ __forceinline__ v16bf load_frag(const bf16_t* base, int pitch) {
  const int l  = threadIdx.x & 31;
  const int m  = l & 15;
  const int kh = l >> 4;
  const bf16_t* p = base + m * pitch + kh * 8;
  union { v16bf v; uint4 q[2]; } f;
  f.q[0] = *reinterpret_cast<const uint4*>(p);       // K = 8*kh .. 8*kh+7
  f.q[1] = *reinterpret_cast<const uint4*>(p + 16);  // K = 16 + 8*kh .. +7
  return f.v;
}

// Stage a 128-row x 32-K f32 tile (row-major source) -> bf16 LDS [row][k].
__device__ __forceinline__ void stage_rows(bf16_t* lds, const float* p, int ld) {
  const int t = threadIdx.x;
#pragma unroll
  for (int it = 0; it < 4; ++it) {
    const int idx = t + it * 256;        // 1024 float4 units
    const int r   = idx >> 3;
    const int c   = (idx & 7) << 2;
    const float4 v = *reinterpret_cast<const float4*>(p + (long)r * ld + c);
    bf16_t* d = lds + r * KP + c;
    d[0] = (bf16_t)v.x; d[1] = (bf16_t)v.y; d[2] = (bf16_t)v.z; d[3] = (bf16_t)v.w;
  }
}

// Stage a 32-K-row x 128-col f32 tile transposed -> bf16 LDS [col][k].
__device__ __forceinline__ void stage_trans128(bf16_t* lds, const float* p, int ld) {
  const int t = threadIdx.x;
#pragma unroll
  for (int it = 0; it < 4; ++it) {
    const int idx = t + it * 256;        // kk in [0,32), n4 in [0,32)
    const int kk  = idx >> 5;
    const int n   = (idx & 31) << 2;
    const float4 v = *reinterpret_cast<const float4*>(p + (long)kk * ld + n);
    lds[(n + 0) * KP + kk] = (bf16_t)v.x;
    lds[(n + 1) * KP + kk] = (bf16_t)v.y;
    lds[(n + 2) * KP + kk] = (bf16_t)v.z;
    lds[(n + 3) * KP + kk] = (bf16_t)v.w;
  }
}

// Stage a 32-K-row x 256-col f32 tile transposed -> bf16 LDS [col][k].
__device__ __forceinline__ void stage_trans256(bf16_t* lds, const float* p, int ld) {
  const int t = threadIdx.x;
#pragma unroll
  for (int it = 0; it < 8; ++it) {
    const int idx = t + it * 256;        // kk in [0,32), n4 in [0,64)
    const int kk  = idx >> 6;
    const int n   = (idx & 63) << 2;
    const float4 v = *reinterpret_cast<const float4*>(p + (long)kk * ld + n);
    lds[(n + 0) * KP + kk] = (bf16_t)v.x;
    lds[(n + 1) * KP + kk] = (bf16_t)v.y;
    lds[(n + 2) * KP + kk] = (bf16_t)v.z;
    lds[(n + 3) * KP + kk] = (bf16_t)v.w;
  }
}

// ---------------------------------------------------------------------------
// Generic 128x256-tile bf16 WMMA GEMM.
//   C_tile = A[MxK] * B[KxN]
//   TRANS_A : A staged transposed (A'[m][k] = A[k][m])
//   ASYNC_B : B source is pre-converted bf16 in [N][K] row-major layout
//             (row pitch ldb = K-extent); staged with CDNA5
//             global_load_async_to_lds_b128 (ASYNCcnt), no VALU conversion.
//   MODE 0  : C = acc + bias[col]     MODE 1 : C += alpha * acc
// grid = (N/256, M/128, batch), 256 threads (8 waves, 4x4 WMMA tiles per wave)
// ---------------------------------------------------------------------------
template <bool TRANS_A, int MODE, bool ASYNC_B>
__global__ void __launch_bounds__(256)
gemm_bf16_kernel(const float* __restrict__ A, long sAb, int lda,
                 const void* __restrict__ Bsv, long sBb, int ldb,
                 float* __restrict__ Cm, long sCb, int ldc,
                 int K,
                 const float* __restrict__ alpha_ptr, float alpha_sign,
                 const float* __restrict__ bias) {
  __shared__ __attribute__((aligned(16))) bf16_t ldsA[128 * KP];
  __shared__ __attribute__((aligned(16))) bf16_t ldsB[256 * KP];

  const int n0 = blockIdx.x * 256;
  const int m0 = blockIdx.y * 128;
  const int bz = blockIdx.z;
  A  += (long)bz * sAb;
  Cm += (long)bz * sCb;
  const float*  Bf = (const float*)Bsv + (long)bz * sBb;   // f32 [K][N]
  const bf16_t* Bh = (const bf16_t*)Bsv + (long)bz * sBb;  // bf16 [N][K]

  const int t    = threadIdx.x;
  const int wave = t >> 5;
  const int wm   = wave >> 2;  // 2 row-groups of 64
  const int wn   = wave & 3;   // 4 col-groups of 64

  v8f acc[4][4];
#pragma unroll
  for (int i = 0; i < 4; ++i)
#pragma unroll
    for (int j = 0; j < 4; ++j) acc[i][j] = v8f_zero();

  for (int k0 = 0; k0 < K; k0 += 32) {
    // ---- stage A (f32 -> bf16, [m][k]) ----
    if (TRANS_A)
      stage_trans128(ldsA, A + (long)k0 * lda + m0, lda);
    else
      stage_rows(ldsA, A + (long)m0 * lda + k0, lda);

    // ---- stage B ----
    if (ASYNC_B) {
      // 256 rows x 64B, pure byte DMA into fragment-layout LDS tile.
#pragma unroll
      for (int it = 0; it < 4; ++it) {
        const int idx = t + it * 256;        // row in [0,256), q in [0,4)
        const int row = idx >> 2;
        const int q   = idx & 3;
        const unsigned goff =
            (unsigned)(((n0 + row) * ldb + k0 + q * 8) * 2);       // bytes
        const unsigned laddr =
            (unsigned)(size_t)(const void*)(ldsB + row * KP + q * 8);
        asm volatile("global_load_async_to_lds_b128 %0, %1, %2"
                     :
                     : "v"(laddr), "v"(goff), "s"(Bh)
                     : "memory");
      }
      asm volatile("s_wait_asynccnt 0x0" ::: "memory");
    } else {
      stage_trans256(ldsB, Bf + (long)k0 * ldb + n0, ldb);
      if (k0 + 32 < K)   // gfx1250 global_prefetch for next K tile
        __builtin_prefetch(Bf + (long)(k0 + 32) * ldb + n0 + (t & 63) * 16, 0, 0);
    }
    __syncthreads();

    // ---- 4x4 WMMA tiles per wave ----
    v16bf af[4];
#pragma unroll
    for (int tm = 0; tm < 4; ++tm)
      af[tm] = load_frag(ldsA + (wm * 64 + tm * 16) * KP, KP);
#pragma unroll
    for (int tn = 0; tn < 4; ++tn) {
      const v16bf bfv = load_frag(ldsB + (wn * 64 + tn * 16) * KP, KP);
#pragma unroll
      for (int tm = 0; tm < 4; ++tm)
        acc[tm][tn] = __builtin_amdgcn_wmma_f32_16x16x32_bf16(
            false, af[tm], false, bfv, (short)0, acc[tm][tn], false, false);
    }
    __syncthreads();
  }

  const float alpha = alpha_sign * (alpha_ptr ? *alpha_ptr : 1.0f);
  const int l   = t & 31;
  const int kh  = l >> 4;
  const int col = l & 15;
#pragma unroll
  for (int tm = 0; tm < 4; ++tm) {
    const int rbase = m0 + wm * 64 + tm * 16 + kh * 8;
#pragma unroll
    for (int tn = 0; tn < 4; ++tn) {
      const int cbase = n0 + wn * 64 + tn * 16 + col;
#pragma unroll
      for (int r = 0; r < 8; ++r) {
        const float v = acc[tm][tn][r];
        float* cp = Cm + (long)(rbase + r) * ldc + cbase;
        if (MODE == 0)
          *cp = v + (bias ? bias[cbase] : 0.0f);
        else
          *cp += alpha * v;
      }
    }
  }
}

// ---------------------------------------------------------------------------
// Intra-chunk causal attention, one workgroup per (b,g):
//   qk = qc @ kc^T (128x128, K=1024), strict-lower mask, requantize to bf16,
//   loc = qk @ vc  (128x1024, K=128),  out -= kstep * loc
// ---------------------------------------------------------------------------
__global__ void __launch_bounds__(256)
intra_chunk_kernel(const float* __restrict__ x, const float* __restrict__ kin,
                   const float* __restrict__ vin, float* __restrict__ out,
                   const float* __restrict__ kstep_ptr) {
  __shared__ __attribute__((aligned(16))) bf16_t ldsA[128 * KP];
  __shared__ __attribute__((aligned(16))) bf16_t ldsB[128 * KP];
  __shared__ __attribute__((aligned(16))) bf16_t ldsQK[128 * QKP];

  const int g = blockIdx.x;
  const int b = blockIdx.y;
  const long row0 = (long)b * L_ + (long)g * C_;
  const float* qbase = x   + row0 * D_;
  const float* kbase = kin + row0 * D_;
  const float* vbase = vin + row0 * S_;
  float*       obase = out + row0 * S_;

  const int wave = threadIdx.x >> 5;
  const int wm   = wave & 3;
  const int wn   = wave >> 2;
  const int l    = threadIdx.x & 31;
  const int kh   = l >> 4;
  const int col  = l & 15;

  // ---- Phase 1: qk = qc @ kc^T  (kc rows already give the [N][K] tile) ----
  v8f acc[2][4];
#pragma unroll
  for (int i = 0; i < 2; ++i)
#pragma unroll
    for (int j = 0; j < 4; ++j) acc[i][j] = v8f_zero();

  for (int k0 = 0; k0 < D_; k0 += 32) {
    stage_rows(ldsA, qbase + k0, D_);
    stage_rows(ldsB, kbase + k0, D_);
    __syncthreads();
    v16bf af[2], bfv[4];
#pragma unroll
    for (int tm = 0; tm < 2; ++tm)
      af[tm] = load_frag(ldsA + (wm * 32 + tm * 16) * KP, KP);
#pragma unroll
    for (int tn = 0; tn < 4; ++tn)
      bfv[tn] = load_frag(ldsB + (wn * 64 + tn * 16) * KP, KP);
#pragma unroll
    for (int tm = 0; tm < 2; ++tm)
#pragma unroll
      for (int tn = 0; tn < 4; ++tn)
        acc[tm][tn] = __builtin_amdgcn_wmma_f32_16x16x32_bf16(
            false, af[tm], false, bfv[tn], (short)0, acc[tm][tn], false, false);
    __syncthreads();
  }

  // strict lower-triangular mask (keep qk[i][j] iff j < i), requantize to bf16
#pragma unroll
  for (int tm = 0; tm < 2; ++tm)
#pragma unroll
    for (int tn = 0; tn < 4; ++tn)
#pragma unroll
      for (int r = 0; r < 8; ++r) {
        const int i = wm * 32 + tm * 16 + kh * 8 + r;
        const int j = wn * 64 + tn * 16 + col;
        const float v = (j < i) ? acc[tm][tn][r] : 0.0f;
        ldsQK[i * QKP + j] = (bf16_t)v;
      }
  __syncthreads();

  const float kstep = *kstep_ptr;

  // ---- Phase 2: loc = qk @ vc, 8 e-tiles of 128 ----
  for (int et = 0; et < 8; ++et) {
    v8f acc2[2][4];
#pragma unroll
    for (int i = 0; i < 2; ++i)
#pragma unroll
      for (int j = 0; j < 4; ++j) acc2[i][j] = v8f_zero();

    for (int ks = 0; ks < 4; ++ks) {
      stage_trans128(ldsB, vbase + (long)(ks * 32) * S_ + et * 128, S_);
      __syncthreads();
      v16bf af[2], bfv[4];
#pragma unroll
      for (int tm = 0; tm < 2; ++tm)
        af[tm] = load_frag(ldsQK + (wm * 32 + tm * 16) * QKP + ks * 32, QKP);
#pragma unroll
      for (int tn = 0; tn < 4; ++tn)
        bfv[tn] = load_frag(ldsB + (wn * 64 + tn * 16) * KP, KP);
#pragma unroll
      for (int tm = 0; tm < 2; ++tm)
#pragma unroll
        for (int tn = 0; tn < 4; ++tn)
          acc2[tm][tn] = __builtin_amdgcn_wmma_f32_16x16x32_bf16(
              false, af[tm], false, bfv[tn], (short)0, acc2[tm][tn], false, false);
      __syncthreads();
    }

#pragma unroll
    for (int tm = 0; tm < 2; ++tm)
#pragma unroll
      for (int tn = 0; tn < 4; ++tn)
#pragma unroll
        for (int r = 0; r < 8; ++r) {
          const int i = wm * 32 + tm * 16 + kh * 8 + r;
          const int e = et * 128 + wn * 64 + tn * 16 + col;
          obase[(long)i * S_ + e] -= kstep * acc2[tm][tn][r];
        }
  }
}

// ---------------------------------------------------------------------------
// Convert W[k][s] (f32) -> Wt[s][k] (bf16, [N][K] layout for async B staging)
// ---------------------------------------------------------------------------
__global__ void convertW_kernel(const float* __restrict__ W, bf16_t* __restrict__ Wt) {
  const long n = (long)blockIdx.x * blockDim.x + threadIdx.x;  // over S_*D_
  const int s = (int)(n / D_);
  const int k = (int)(n % D_);
  Wt[n] = (bf16_t)W[(long)k * S_ + s];
}

// ---------------------------------------------------------------------------
// Exclusive cumsum of upstream_grad over L, chunked 3-pass scan.
// ---------------------------------------------------------------------------
__global__ void chunk_sum_kernel(const float* __restrict__ ug, float* __restrict__ csum) {
  const int s = blockIdx.x * 256 + threadIdx.x;
  const int g = blockIdx.y, b = blockIdx.z;
  const float* p = ug + ((long)b * L_ + (long)g * C_) * S_ + s;
  float acc = 0.f;
  for (int i = 0; i < C_; ++i) acc += p[(long)i * S_];
  csum[((long)b * G_ + g) * S_ + s] = acc;
}

__global__ void chunk_prefix_kernel(float* __restrict__ csum) {
  const int s = blockIdx.x * 256 + threadIdx.x;
  const int b = blockIdx.y;
  float* p = csum + (long)b * G_ * S_ + s;
  float acc = 0.f;
  for (int gg = 0; gg < G_; ++gg) {
    const float t = p[(long)gg * S_];
    p[(long)gg * S_] = acc;
    acc += t;
  }
}

__global__ void bias_apply_kernel(const float* __restrict__ ug,
                                  const float* __restrict__ csum,
                                  float* __restrict__ out,
                                  const float* __restrict__ bstep_ptr) {
  const int s = blockIdx.x * 256 + threadIdx.x;
  const int g = blockIdx.y, b = blockIdx.z;
  const float bstep = *bstep_ptr;
  const float* p = ug  + ((long)b * L_ + (long)g * C_) * S_ + s;
  float*       o = out + ((long)b * L_ + (long)g * C_) * S_ + s;
  float acc = csum[((long)b * G_ + g) * S_ + s];
  for (int i = 0; i < C_; ++i) {
    o[(long)i * S_] -= bstep * acc;    // exclusive
    acc += p[(long)i * S_];
  }
}

__global__ void zero_kernel(float* __restrict__ p, long n) {
  const long i = (long)blockIdx.x * blockDim.x + threadIdx.x;
  if (i < n) p[i] = 0.f;
}

// ---------------------------------------------------------------------------
extern "C" void kernel_launch(void* const* d_in, const int* in_sizes, int n_in,
                              void* d_out, int out_size, void* d_ws, size_t ws_size,
                              hipStream_t stream) {
  (void)in_sizes; (void)n_in; (void)out_size; (void)ws_size;
  const float* x    = (const float*)d_in[0];
  const float* sw   = (const float*)d_in[1];
  const float* ug   = (const float*)d_in[2];
  const float* W    = (const float*)d_in[3];
  const float* bias = (const float*)d_in[4];
  const float* kst  = (const float*)d_in[5];
  const float* bst  = (const float*)d_in[6];
  float* out = (float*)d_out;

  float*  kvstate = (float*)d_ws;                          // B*S*S f32 (16 MB)
  float*  csum    = kvstate + (long)B_ * S_ * S_;          // B*G*S f32 (512 KB)
  bf16_t* Wt      = (bf16_t*)(csum + (long)B_ * G_ * S_);  // S*D bf16 (2 MB)

  // KV running state must start at zero every call (exclusive cumsum).
  const long nz = (long)B_ * S_ * S_;
  zero_kernel<<<dim3((unsigned)((nz + 255) / 256)), 256, 0, stream>>>(kvstate, nz);

  // Pre-convert/transpose weights once: enables async bf16 B staging in K1.
  convertW_kernel<<<dim3((S_ * D_) / 256), 256, 0, stream>>>(W, Wt);

  // out = x @ W + bias   (M = B*L flat, N = S, K = D) — async-B path
  gemm_bf16_kernel<false, 0, true><<<dim3(S_ / 256, (B_ * L_) / 128, 1), 256, 0, stream>>>(
      x, 0, D_, Wt, 0, D_, out, 0, S_, D_, nullptr, 1.0f, bias);

  // Inter-chunk linear attention: serial recurrence over chunks.
  for (int g = 0; g < G_; ++g) {
    // glob: out[b, chunk g] -= kstep * qc @ KVstate[b]   (exclusive: read first)
    gemm_bf16_kernel<false, 1, false><<<dim3(S_ / 256, 1, B_), 256, 0, stream>>>(
        x + (long)g * C_ * D_, (long)L_ * D_, D_,
        kvstate, (long)S_ * S_, S_,
        out + (long)g * C_ * S_, (long)L_ * S_, S_,
        D_, kst, -1.0f, nullptr);
    // KVstate[b] += kc^T @ vc
    gemm_bf16_kernel<true, 1, false><<<dim3(S_ / 256, D_ / 128, B_), 256, 0, stream>>>(
        sw + (long)g * C_ * D_, (long)L_ * D_, D_,
        ug + (long)g * C_ * S_, (long)L_ * S_, S_,
        kvstate, (long)S_ * S_, S_,
        C_, nullptr, 1.0f, nullptr);
  }

  // Intra-chunk quadratic causal attention.
  intra_chunk_kernel<<<dim3(G_, B_), 256, 0, stream>>>(x, sw, ug, out, kst);

  // Exclusive cumsum bias term.
  chunk_sum_kernel<<<dim3(S_ / 256, G_, B_), 256, 0, stream>>>(ug, csum);
  chunk_prefix_kernel<<<dim3(S_ / 256, B_), 256, 0, stream>>>(csum);
  bias_apply_kernel<<<dim3(S_ / 256, G_, B_), 256, 0, stream>>>(ug, csum, out, bst);
}
